// snn_n_model_80032420593679
// MI455X (gfx1250) — matrix operand order repaired and checked by reading, compiled, not run on hardware
//
#include <hip/hip_runtime.h>

// ---------------------------------------------------------------------------
// SNN recurrence on MI455X (gfx1250):
//   persistent kernel; membranes pinned in WMMA-D-layout VGPRs for 512 steps;
//   f16 WMMA (f32 accum); weights pre-swizzled to B-operand layout and parked
//   in 128 KB of LDS per WG for the whole kernel (320 KB/WGP on CDNA5);
//   spikes exchanged via small L2-resident f16 buffers; epoch grid barriers.
// ---------------------------------------------------------------------------

typedef _Float16 h16;
typedef __attribute__((ext_vector_type(16))) _Float16 v16h;
typedef __attribute__((ext_vector_type(8)))  _Float16 v8h;
typedef __attribute__((ext_vector_type(8)))  float    v8f;

#define SNN_THRESH 0.3f
#define T_STEPS 512
#define BATCH   128
#define D0      512
#define D1      1024
#define D2      1024
#define D3      512

#define NWG   32
#define NTHR  256   // 8 wave32 per WG

// workspace layout (bytes)
#define WS_BAR   0
#define WS_XBUF  256                        // 128x512  f16 = 131072
#define WS_S0    (WS_XBUF + BATCH*D0*2)     // 128x1024 f16 = 262144
#define WS_S1    (WS_S0   + BATCH*D1*2)     // 128x1024 f16 = 262144
#define WS_W0T   (WS_S1   + BATCH*D2*2)     // 512x1024 f16 = 1 MB
#define WS_W1T   (WS_W0T  + D0*D1*2)        // 1024x1024 f16 = 2 MB
#define WS_W2T   (WS_W1T  + D1*D2*2)        // 1024x512 f16 = 1 MB

// LDS layout (h16 indices): per-WG weight slices, resident for whole kernel
#define LDS_W0   0                          // 2 strips * 16 kt * 512 = 16384
#define LDS_W1   16384                      // 2 strips * 32 kt * 512 = 32768
#define LDS_W2   49152                      // 1 strip  * 32 kt * 512 = 16384
#define LDS_H16  65536                      // 128 KB total

__device__ inline v8f vzero8() {
  v8f v;
#pragma unroll
  for (int i = 0; i < 8; ++i) v[i] = 0.0f;
  return v;
}

// --------------------------- weight pre-swizzle ----------------------------
// B-operand tile layout for v_wmma_*_16x16x32 (32x16, 16-bit):
//   block (nt,kt) = 32 lanes x 16 f16 contiguous;
//   lane = n + 16*kh : column N = nt*16+n, K = kt*32 + kh*8 + (i&7) + (i>>3)*16 (i=0..15)
// Strips (all kt for one nt) are contiguous: strip nt = [nt*KT*512, ...)
__global__ void convert_weights(const float* __restrict__ W, h16* __restrict__ Wt,
                                int K, int N) {
  int KT = K >> 5;
  int total = K * N;
  int e = blockIdx.x * blockDim.x + threadIdx.x;
  if (e >= total) return;
  int blk  = e >> 9;          // 512 elements per 32x16 block
  int r    = e & 511;
  int lane = r >> 4;
  int i    = r & 15;
  int nt   = blk / KT;
  int kt   = blk - nt * KT;
  int n    = nt * 16 + (lane & 15);
  int k    = kt * 32 + ((lane >> 4) << 3) + (i & 7) + ((i >> 3) << 4);
  Wt[e] = (h16)W[(size_t)k * N + n];
}

__global__ void init_barrier(unsigned* bar) {
  if (threadIdx.x < 2) bar[threadIdx.x] = 0u;  // [0]=count, [1]=epoch sense
}

// ------------------------------ grid barrier -------------------------------
__device__ inline void grid_barrier(unsigned* cnt, unsigned* sense, unsigned* epoch) {
  __threadfence();
  __syncthreads();
  if (threadIdx.x == 0) {
    unsigned target = *epoch + 1u;
    unsigned arrived =
        __hip_atomic_fetch_add(cnt, 1u, __ATOMIC_ACQ_REL, __HIP_MEMORY_SCOPE_AGENT) + 1u;
    if (arrived == NWG) {
      __hip_atomic_store(cnt, 0u, __ATOMIC_RELAXED, __HIP_MEMORY_SCOPE_AGENT);
      __hip_atomic_store(sense, target, __ATOMIC_RELEASE, __HIP_MEMORY_SCOPE_AGENT);
    } else {
      while (__hip_atomic_load(sense, __ATOMIC_ACQUIRE, __HIP_MEMORY_SCOPE_AGENT) != target)
        __builtin_amdgcn_s_sleep(1);
    }
  }
  *epoch += 1u;
  __syncthreads();
  __threadfence();
}

// ------------------------------ WMMA helpers -------------------------------
__device__ inline v8f wmma16(v16h a, v16h b, v8f c) {
  // D = A(16x32,f16) x B(32x16,f16) + C(16x16,f32)
  return __builtin_amdgcn_wmma_f32_16x16x32_f16(false, a, false, b, (short)0, c,
                                                false, false);
}

// A operand (16x32 f16) from row-major S[.,ldS] in global memory:
//   lane<16  : row = mt*16+lane,    K = kt*32 + {0..7, 16..23}
//   lane>=16 : row = mt*16+lane-16, K = kt*32 + {8..15, 24..31}
__device__ inline v16h load_a(const h16* __restrict__ S, int ldS, int mt, int kt,
                              int lane) {
  int row = mt * 16 + (lane & 15);
  int k0  = kt * 32 + ((lane >> 4) << 3);
  const h16* p = S + (size_t)row * ldS + k0;
  v8h lo = *(const v8h*)(p);
  v8h hi = *(const v8h*)(p + 16);
  v16h a;
#pragma unroll
  for (int i = 0; i < 8; ++i) { a[i] = lo[i]; a[i + 8] = hi[i]; }
  return a;
}

// epilogues: mem = mem*0.5*(1-s_old) + z ; s_old = mem_old > TH ; fire spikes
__device__ inline void update_fire_spk(v8f& mem, v8f z, h16* __restrict__ sbuf,
                                       int ldN, int mt, int nt, int lane) {
  int col   = nt * 16 + (lane & 15);
  int rbase = mt * 16 + ((lane >> 4) << 3);   // D-matrix layout rows
#pragma unroll
  for (int r = 0; r < 8; ++r) {
    float m = mem[r];
    m = ((m > SNN_THRESH) ? 0.0f : m * 0.5f) + z[r];
    mem[r] = m;
    sbuf[(size_t)(rbase + r) * ldN + col] = (m > SNN_THRESH) ? (h16)1.0f : (h16)0.0f;
  }
}

__device__ inline void update_fire_out(v8f& mem, v8f z, float* __restrict__ fout,
                                       int fld, int mt, int nt, int lane) {
  int col   = nt * 16 + (lane & 15);
  int rbase = mt * 16 + ((lane >> 4) << 3);
#pragma unroll
  for (int r = 0; r < 8; ++r) {
    float m = mem[r];
    m = ((m > SNN_THRESH) ? 0.0f : m * 0.5f) + z[r];
    mem[r] = m;
    // overwrite every step; final replayed value is the t = T-1 spike
    fout[(size_t)(rbase + r) * fld + col] = (m > SNN_THRESH) ? 1.0f : 0.0f;
  }
}

// two column-adjacent 16x16 tiles sharing the A operand; B strips live in LDS
template <int KT>
__device__ inline void layer_pair_lds(const h16* __restrict__ S, int ldS,
                                      const h16* Blds, int mt, int nt0,
                                      v8f& mem0, v8f& mem1, h16* __restrict__ sbuf,
                                      int ldN, int lane) {
  v8f acc0 = vzero8(), acc1 = vzero8();
  const h16* b0p = Blds + lane * 16;              // strip nt0
  const h16* b1p = Blds + KT * 512 + lane * 16;   // strip nt0+1
#pragma unroll 4
  for (int kt = 0; kt < KT; ++kt) {
    v16h a  = load_a(S, ldS, mt, kt, lane);
    v16h b0 = *(const v16h*)(b0p + kt * 512);     // ds_load_b128 x2
    v16h b1 = *(const v16h*)(b1p + kt * 512);
    acc0 = wmma16(a, b0, acc0);
    acc1 = wmma16(a, b1, acc1);
  }
  update_fire_spk(mem0, acc0, sbuf, ldN, mt, nt0, lane);
  update_fire_spk(mem1, acc1, sbuf, ldN, mt, nt0 + 1, lane);
}

template <int KT>
__device__ inline void layer_single_lds(const h16* __restrict__ S, int ldS,
                                        const h16* Blds, int mt, int nt,
                                        v8f& mem, int lane,
                                        float* __restrict__ fout, int fld) {
  v8f acc = vzero8();
  const h16* bp = Blds + lane * 16;
#pragma unroll 4
  for (int kt = 0; kt < KT; ++kt) {
    v16h a = load_a(S, ldS, mt, kt, lane);
    v16h b = *(const v16h*)(bp + kt * 512);
    acc = wmma16(a, b, acc);
  }
  update_fire_out(mem, acc, fout, fld, mt, nt, lane);
}

__device__ inline void copy_to_lds(h16* dst, const h16* __restrict__ src, int n,
                                   int tid) {
  for (int i = tid * 8; i < n; i += NTHR * 8)
    *(v8h*)(dst + i) = *(const v8h*)(src + i);   // global_load_b128 -> ds_store_b128
}

// ----------------------------- persistent kernel ---------------------------
__global__ __launch_bounds__(NTHR)
void snn_persistent(const float* __restrict__ x,
                    const h16* __restrict__ w0t, const h16* __restrict__ w1t,
                    const h16* __restrict__ w2t, h16* __restrict__ xbuf,
                    h16* __restrict__ s0, h16* __restrict__ s1,
                    float* __restrict__ out, unsigned* __restrict__ bar) {
  extern __shared__ h16 ldsw[];                 // 128 KB dynamic LDS
  unsigned* cnt   = bar;
  unsigned* sense = bar + 1;
  unsigned  epoch = 0;

  const int g    = blockIdx.x;                  // WG owns column strips
  const int lane = threadIdx.x & 31;
  const int tid  = threadIdx.x;
  const int mt   = tid >> 5;                    // wave id 0..7 = row tile
  const int gtid = g * NTHR + tid;

  // park this WG's weight slices in LDS for the whole kernel:
  //  w0 strips {2g,2g+1} (32 KB), w1 strips {2g,2g+1} (64 KB), w2 strip g (32 KB)
  copy_to_lds(ldsw + LDS_W0, w0t + (size_t)(2 * g) * (D0 / 32) * 512, 2 * (D0 / 32) * 512, tid);
  copy_to_lds(ldsw + LDS_W1, w1t + (size_t)(2 * g) * (D1 / 32) * 512, 2 * (D1 / 32) * 512, tid);
  copy_to_lds(ldsw + LDS_W2, w2t + (size_t)g * (D2 / 32) * 512, (D2 / 32) * 512, tid);
  __syncthreads();

  const int nt0 = 2 * g;                        // L0/L1 column strips
  const int nt2 = g;                            // L2 column strip

  // membrane state pinned in VGPRs (WMMA C/D layout) for all 512 steps
  v8f m0a = vzero8(), m0b = vzero8();
  v8f m1a = vzero8(), m1b = vzero8();
  v8f m2  = vzero8();

  for (int t = 0; t < T_STEPS; ++t) {
    // Phase X: x[:, :, t] (t-minor strided fp32) -> row-major f16 xbuf[128][512]
    // prefetch 16 steps ahead: one 64B line of x[b,d,:] covers 16 timesteps
    int tp = (t + 16 < T_STEPS) ? t + 16 : t;
    for (int e = gtid; e < BATCH * D0; e += NWG * NTHR) {
      size_t base = (size_t)(e >> 9) * (D0 * T_STEPS) + (size_t)(e & (D0 - 1)) * T_STEPS;
      xbuf[e] = (h16)x[base + t];
      __builtin_prefetch(&x[base + tp], 0, 1);  // global_prefetch_b8
    }
    grid_barrier(cnt, sense, &epoch);

    // Layer 0: xbuf @ w0 -> mem0, fire s0
    layer_pair_lds<D0 / 32>(xbuf, D0, ldsw + LDS_W0, mt, nt0, m0a, m0b, s0, D1, lane);
    grid_barrier(cnt, sense, &epoch);

    // Layer 1: s0 @ w1 -> mem1, fire s1
    layer_pair_lds<D1 / 32>(s0, D1, ldsw + LDS_W1, mt, nt0, m1a, m1b, s1, D2, lane);
    grid_barrier(cnt, sense, &epoch);

    // Layer 2: s1 @ w2 -> mem2, spikes straight to d_out (fp32, overwritten per step)
    layer_single_lds<D2 / 32>(s1, D2, ldsw + LDS_W2, mt, nt2, m2, lane, out, D3);
    grid_barrier(cnt, sense, &epoch);
  }
}

// ------------------------------- launcher ----------------------------------
extern "C" void kernel_launch(void* const* d_in, const int* in_sizes, int n_in,
                              void* d_out, int out_size, void* d_ws, size_t ws_size,
                              hipStream_t stream) {
  const float* x  = (const float*)d_in[0];  // [128,512,512] fp32
  const float* w0 = (const float*)d_in[1];  // [512,1024]
  const float* w1 = (const float*)d_in[2];  // [1024,1024]
  const float* w2 = (const float*)d_in[3];  // [1024,512]

  char* ws = (char*)d_ws;                   // needs ~4.8 MB
  unsigned* bar = (unsigned*)(ws + WS_BAR);
  h16* xbuf = (h16*)(ws + WS_XBUF);
  h16* s0   = (h16*)(ws + WS_S0);
  h16* s1   = (h16*)(ws + WS_S1);
  h16* w0t  = (h16*)(ws + WS_W0T);
  h16* w1t  = (h16*)(ws + WS_W1T);
  h16* w2t  = (h16*)(ws + WS_W2T);

  init_barrier<<<1, 32, 0, stream>>>(bar);
  convert_weights<<<(D0 * D1 + 255) / 256, 256, 0, stream>>>(w0, w0t, D0, D1);
  convert_weights<<<(D1 * D2 + 255) / 256, 256, 0, stream>>>(w1, w1t, D1, D2);
  convert_weights<<<(D2 * D3 + 255) / 256, 256, 0, stream>>>(w2, w2t, D2, D3);

  snn_persistent<<<NWG, NTHR, LDS_H16 * sizeof(h16), stream>>>(
      x, w0t, w1t, w2t, xbuf, s0, s1, (float*)d_out, bar);
}
